// DiagonnalyMaskedSelfAttention_2491081032086
// MI455X (gfx1250) — matrix operand order repaired and checked by reading, not compiled, not run on hardware
//
#include <hip/hip_runtime.h>
#include <cstdint>
#include <cstddef>

// ---------------------------------------------------------------------------
// DiagonallyMaskedSelfAttention for MI455X (gfx1250, wave32, WMMA f16->f32)
// B=2, L=2048, DIM=1024, H=16, D=64
// ---------------------------------------------------------------------------

typedef __attribute__((ext_vector_type(16))) _Float16 v16h;
typedef __attribute__((ext_vector_type(8)))  _Float16 v8h;
typedef __attribute__((ext_vector_type(4)))  _Float16 v4h;
typedef __attribute__((ext_vector_type(8)))  float    v8f;

#define DEVFN static __device__ __forceinline__

constexpr int Bc   = 2;
constexpr int Lc   = 2048;
constexpr int DIMc = 1024;
constexpr int Hc   = 16;
constexpr int Dc   = 64;

DEVFN v16h make_v16(v8h lo, v8h hi) {
  return __builtin_shufflevector(lo, hi, 0,1,2,3,4,5,6,7,8,9,10,11,12,13,14,15);
}

DEVFN v8f wmma_f16(v16h a, v16h b, v8f c) {
  // v_wmma_f32_16x16x32_f16  (neg_a, A, neg_b, B, c_mod, C, reuse_a, reuse_b)
  return __builtin_amdgcn_wmma_f32_16x16x32_f16(false, a, false, b, (short)0, c,
                                                false, false);
}

DEVFN v8f vzero8() { v8f z = {0.f,0.f,0.f,0.f,0.f,0.f,0.f,0.f}; return z; }

// ---------------------------------------------------------------------------
// f32 -> f16 elementwise convert (4 elements / thread, n multiple of 4)
// ---------------------------------------------------------------------------
__global__ __launch_bounds__(256) void cvt_f32_f16(const float* __restrict__ in,
                                                   _Float16* __restrict__ out) {
  size_t i = ((size_t)blockIdx.x * blockDim.x + threadIdx.x) * 4;
  float4 f = *(const float4*)(in + i);
  v4h h;
  h.x = (_Float16)f.x; h.y = (_Float16)f.y;
  h.z = (_Float16)f.z; h.w = (_Float16)f.w;
  *(v4h*)(out + i) = h;
}

// ---------------------------------------------------------------------------
// f32 [1024][1024] -> f16 transposed Wt[n][k] = W[k][n]   (K-contiguous for B)
// ---------------------------------------------------------------------------
__global__ __launch_bounds__(256) void cvt_w_transpose(const float* __restrict__ W,
                                                       _Float16* __restrict__ Wt) {
  __shared__ float tile[32][33];
  const int tx = threadIdx.x & 31, ty = threadIdx.x >> 5;
  const int k0 = blockIdx.x * 32, n0 = blockIdx.y * 32;
#pragma unroll
  for (int i = 0; i < 4; ++i)
    tile[ty + 8*i][tx] = W[(size_t)(k0 + ty + 8*i) * DIMc + n0 + tx];
  __syncthreads();
#pragma unroll
  for (int i = 0; i < 4; ++i)
    Wt[(size_t)(n0 + ty + 8*i) * DIMc + k0 + tx] = (_Float16)tile[tx][ty + 8*i];
}

// ---------------------------------------------------------------------------
// WMMA GEMM: C[M=4096][N=1024] = A[M][K=1024] (row-major f16) @ Bt[N][K] (f16)
// 256 threads = 8 waves; wave tile 16x64; block tile 64x128.
// mode 0: write f16 to [B][H][L][D]  (Q / K layout)
// mode 1: write f16 to [B][H][D][L]  (V transposed)
// mode 2: write f32 row-major [M][N] (final output)
// ---------------------------------------------------------------------------
__global__ __launch_bounds__(256)
void gemm_wmma(const _Float16* __restrict__ A, const _Float16* __restrict__ Bt,
               _Float16* __restrict__ Ch, float* __restrict__ Cf, int mode) {
  constexpr int K = DIMc;
  const int lane = threadIdx.x & 31;
  const int w    = threadIdx.x >> 5;
  const int half = lane >> 4, ln = lane & 15;
  const int mbase = blockIdx.x * 64  + (w & 3) * 16;
  const int nbase = blockIdx.y * 128 + (w >> 2) * 64;

  v8f acc0 = vzero8(), acc1 = vzero8(), acc2 = vzero8(), acc3 = vzero8();

  const _Float16* __restrict__ arow = A + (size_t)(mbase + ln) * K;
  const _Float16* __restrict__ b0   = Bt + (size_t)(nbase +  0 + ln) * K + half * 16;
  const _Float16* __restrict__ b1   = Bt + (size_t)(nbase + 16 + ln) * K + half * 16;
  const _Float16* __restrict__ b2   = Bt + (size_t)(nbase + 32 + ln) * K + half * 16;
  const _Float16* __restrict__ b3   = Bt + (size_t)(nbase + 48 + ln) * K + half * 16;

  for (int kk = 0; kk < K; kk += 32) {
    // A fragment: element e -> K = half*16 + e
    v8h alo = *(const v8h*)(arow + kk + half * 16);
    v8h ahi = *(const v8h*)(arow + kk + half * 16 + 8);
    v16h a  = make_v16(alo, ahi);
    // B fragments: element e -> K = half*16 + e (contiguous 32B)
    v16h f0 = *(const v16h*)(b0 + kk);
    v16h f1 = *(const v16h*)(b1 + kk);
    v16h f2 = *(const v16h*)(b2 + kk);
    v16h f3 = *(const v16h*)(b3 + kk);
    acc0 = wmma_f16(a, f0, acc0);
    acc1 = wmma_f16(a, f1, acc1);
    acc2 = wmma_f16(a, f2, acc2);
    acc3 = wmma_f16(a, f3, acc3);
  }

#pragma unroll
  for (int nb = 0; nb < 4; ++nb) {
    const v8f acc = (nb == 0) ? acc0 : (nb == 1) ? acc1 : (nb == 2) ? acc2 : acc3;
#pragma unroll
    for (int r = 0; r < 8; ++r) {
      const int m = mbase + r + 8 * half;            // C row  = r + 8*half
      const int n = nbase + nb * 16 + ln;            // C col  = lane%16
      const float v = acc[r];
      if (mode == 2) {
        Cf[(size_t)m * DIMc + n] = v;
      } else {
        const int b = m >> 11, l = m & (Lc - 1);
        const int h = n >> 6,  d = n & (Dc - 1);
        const _Float16 hv = (_Float16)v;
        if (mode == 0)
          Ch[(((size_t)(b * Hc + h)) * Lc + l) * Dc + d] = hv;   // [B][H][L][D]
        else
          Ch[(((size_t)(b * Hc + h)) * Dc + d) * Lc + l] = hv;   // [B][H][D][L]
      }
    }
  }
}

// ---------------------------------------------------------------------------
// Flash-style attention with diagonal mask (q == k forbidden).
// grid = (L/128, B*H); 256 threads = 8 waves; wave handles 16 q rows.
// kv processed in chunks of 32; online softmax in C-fragment layout.
// ---------------------------------------------------------------------------
__global__ __launch_bounds__(256)
void attn_wmma(const _Float16* __restrict__ Qb, const _Float16* __restrict__ Kb,
               const _Float16* __restrict__ Vt, _Float16* __restrict__ O16) {
  __shared__ _Float16 pt[8][16][32];   // per-wave P tile (C-layout -> A-layout)

  const float scale = 0.125f;          // 1/sqrt(64)
  const int lane = threadIdx.x & 31;
  const int w    = threadIdx.x >> 5;
  const int half = lane >> 4, ln = lane & 15;
  const int bh   = blockIdx.y;                 // b*H + h
  const int qb   = blockIdx.x * 128 + w * 16;  // q tile base in [0, L)

  const _Float16* __restrict__ Qh = Qb + (size_t)bh * Lc * Dc;
  const _Float16* __restrict__ Kh = Kb + (size_t)bh * Lc * Dc;
  const _Float16* __restrict__ Vh = Vt + (size_t)bh * Dc * Lc;

  // Q fragments (held for the whole kv loop): d in [0,32) and [32,64)
  const _Float16* __restrict__ qrow = Qh + (size_t)(qb + ln) * Dc;
  const v16h aq0 = make_v16(*(const v8h*)(qrow +      half * 16),
                            *(const v8h*)(qrow +  8 + half * 16));
  const v16h aq1 = make_v16(*(const v8h*)(qrow + 32 + half * 16),
                            *(const v8h*)(qrow + 40 + half * 16));

  v8f o0 = vzero8(), o1 = vzero8(), o2 = vzero8(), o3 = vzero8();
  float mrow[8], srow[8];
#pragma unroll
  for (int r = 0; r < 8; ++r) { mrow[r] = -__builtin_inff(); srow[r] = 0.f; }

  for (int jb = 0; jb < Lc; jb += 32) {
    // ---- S = Q . K^T over this 32-wide kv chunk (two 16-col C fragments) ---
    const _Float16* k0p = Kh + (size_t)(jb +      ln) * Dc + half * 16;
    const _Float16* k1p = Kh + (size_t)(jb + 16 + ln) * Dc + half * 16;
    __builtin_prefetch(k0p + 32 * Dc, 0, 3);     // next chunk -> L2/L0
    v16h bk00 = *(const v16h*)(k0p);             // d 0..31
    v16h bk01 = *(const v16h*)(k0p + 32);        // d 32..63
    v16h bk10 = *(const v16h*)(k1p);
    v16h bk11 = *(const v16h*)(k1p + 32);

    v8f s0 = vzero8(), s1 = vzero8();
    s0 = wmma_f16(aq0, bk00, s0);
    s0 = wmma_f16(aq1, bk01, s0);
    s1 = wmma_f16(aq0, bk10, s1);
    s1 = wmma_f16(aq1, bk11, s1);

    // ---- diagonal mask + online softmax (row stats per VGPR r) ------------
    float alpha[8];
#pragma unroll
    for (int r = 0; r < 8; ++r) {
      const int qidx = qb + r + 8 * half;
      float t0 = s0[r] * scale; if (qidx == jb + ln)      t0 = -__builtin_inff();
      float t1 = s1[r] * scale; if (qidx == jb + 16 + ln) t1 = -__builtin_inff();
      float cm = fmaxf(t0, t1);
      cm = fmaxf(cm, __shfl_xor(cm, 1, 32));
      cm = fmaxf(cm, __shfl_xor(cm, 2, 32));
      cm = fmaxf(cm, __shfl_xor(cm, 4, 32));
      cm = fmaxf(cm, __shfl_xor(cm, 8, 32));
      const float mn = fmaxf(mrow[r], cm);       // always finite (<=1 masked col)
      const float a  = __expf(mrow[r] - mn);
      const float e0 = __expf(t0 - mn);
      const float e1 = __expf(t1 - mn);
      float rs = e0 + e1;
      rs += __shfl_xor(rs, 1, 32);
      rs += __shfl_xor(rs, 2, 32);
      rs += __shfl_xor(rs, 4, 32);
      rs += __shfl_xor(rs, 8, 32);
      srow[r] = srow[r] * a + rs;
      mrow[r] = mn;
      alpha[r] = a;
      // stage P (f16) into LDS in C layout
      pt[w][r + 8 * half][ln]      = (_Float16)e0;
      pt[w][r + 8 * half][ln + 16] = (_Float16)e1;
    }
#pragma unroll
    for (int r = 0; r < 8; ++r) {
      o0[r] *= alpha[r]; o1[r] *= alpha[r];
      o2[r] *= alpha[r]; o3[r] *= alpha[r];
    }

    __syncthreads();   // C-layout stores visible before A-layout reads

    // re-read P as an A fragment: row = lane%16, K = half*16 + e
    const _Float16* lp = &pt[w][ln][half * 16];
    v16h ap = make_v16(*(const v8h*)(lp), *(const v8h*)(lp + 8));

    // ---- O += P . V  (V stored transposed: [D][L], l-contiguous) ----------
    const _Float16* vp = Vh + (size_t)ln * Lc + jb + half * 16;
    __builtin_prefetch(vp + 32, 0, 3);
    v16h bv0 = *(const v16h*)(vp);
    v16h bv1 = *(const v16h*)(vp + 16 * Lc);
    v16h bv2 = *(const v16h*)(vp + 32 * Lc);
    v16h bv3 = *(const v16h*)(vp + 48 * Lc);
    o0 = wmma_f16(ap, bv0, o0);
    o1 = wmma_f16(ap, bv1, o1);
    o2 = wmma_f16(ap, bv2, o2);
    o3 = wmma_f16(ap, bv3, o3);

    __syncthreads();   // loads done before next iteration overwrites pt
  }

  // ---- normalize and store O as f16 row-major [B*L][DIM] -------------------
  const int b = bh >> 4, h = bh & 15;
#pragma unroll
  for (int r = 0; r < 8; ++r) {
    const float inv = 1.f / srow[r];
    const int q = qb + r + 8 * half;
    _Float16* op = O16 + ((size_t)(b * Lc + q)) * DIMc + h * Dc + ln;
    op[0]  = (_Float16)(o0[r] * inv);
    op[16] = (_Float16)(o1[r] * inv);
    op[32] = (_Float16)(o2[r] * inv);
    op[48] = (_Float16)(o3[r] * inv);
  }
}

// ---------------------------------------------------------------------------
extern "C" void kernel_launch(void* const* d_in, const int* in_sizes, int n_in,
                              void* d_out, int out_size, void* d_ws, size_t ws_size,
                              hipStream_t stream) {
  (void)in_sizes; (void)n_in; (void)out_size; (void)ws_size;
  const float* x  = (const float*)d_in[0];
  const float* Wq = (const float*)d_in[1];
  const float* Wk = (const float*)d_in[2];
  const float* Wv = (const float*)d_in[3];
  const float* Wo = (const float*)d_in[4];
  float* out = (float*)d_out;

  // workspace carve-out (f16 elements), 48 MiB total
  _Float16* p   = (_Float16*)d_ws;
  _Float16* x16 = p; p += (size_t)Bc * Lc * DIMc;        // 8 MiB
  _Float16* Wqt = p; p += (size_t)DIMc * DIMc;           // 2 MiB each
  _Float16* Wkt = p; p += (size_t)DIMc * DIMc;
  _Float16* Wvt = p; p += (size_t)DIMc * DIMc;
  _Float16* Wot = p; p += (size_t)DIMc * DIMc;
  _Float16* Qb  = p; p += (size_t)Bc * Hc * Lc * Dc;     // [B][H][L][D]
  _Float16* Kb  = p; p += (size_t)Bc * Hc * Lc * Dc;
  _Float16* Vtb = p; p += (size_t)Bc * Hc * Dc * Lc;     // [B][H][D][L]
  _Float16* O16 = p; p += (size_t)Bc * Lc * DIMc;

  // 1) precision conversion (+ weight transpose for K-contiguous B operands)
  cvt_f32_f16<<<dim3((Bc * Lc * DIMc) / (256 * 4)), dim3(256), 0, stream>>>(x, x16);
  dim3 tg(DIMc / 32, DIMc / 32);
  cvt_w_transpose<<<tg, dim3(256), 0, stream>>>(Wq, Wqt);
  cvt_w_transpose<<<tg, dim3(256), 0, stream>>>(Wk, Wkt);
  cvt_w_transpose<<<tg, dim3(256), 0, stream>>>(Wv, Wvt);
  cvt_w_transpose<<<tg, dim3(256), 0, stream>>>(Wo, Wot);

  // 2) QKV projections (WMMA)
  dim3 gg((Bc * Lc) / 64, DIMc / 128);
  gemm_wmma<<<gg, dim3(256), 0, stream>>>(x16, Wqt, Qb,  nullptr, 0);
  gemm_wmma<<<gg, dim3(256), 0, stream>>>(x16, Wkt, Kb,  nullptr, 0);
  gemm_wmma<<<gg, dim3(256), 0, stream>>>(x16, Wvt, Vtb, nullptr, 1);

  // 3) diagonally-masked attention (WMMA + online softmax)
  dim3 ag(Lc / 128, Bc * Hc);
  attn_wmma<<<ag, dim3(256), 0, stream>>>(Qb, Kb, Vtb, O16);

  // 4) output projection -> f32 d_out
  gemm_wmma<<<gg, dim3(256), 0, stream>>>(O16, Wot, nullptr, out, 2);
}
